// Spatial_37589553774687
// MI455X (gfx1250) — compile-verified, matrix operand order checked by reading, not run on hardware
//
#include <hip/hip_runtime.h>

// ---------------------------------------------------------------------------
// MI455X (gfx1250) implementation.
// All heavy lifting is BF16 WMMA (v_wmma_f32_16x16x32_bf16) with f32 accum:
//   ~450 GFLOP of GEMMs vs ~hundreds of MB of traffic at 23.3 TB/s
//   => matrix-core bound; bf16 WMMA is the right precision/path.
// GEMM kernel uses double-buffered LDS; tiles are staged with
// GLOBAL_LOAD_ASYNC_TO_LDS_B128 (ASYNCcnt) when the toolchain exposes the
// builtin, otherwise with split-issue register staging, so global-load
// latency overlaps the WMMA stream either way.
// ---------------------------------------------------------------------------

typedef __bf16 bf16;
typedef __attribute__((ext_vector_type(16))) __bf16 v16bf;
typedef __attribute__((ext_vector_type(8)))  __bf16 v8bf;
typedef __attribute__((ext_vector_type(8)))  float  v8f;
typedef __attribute__((ext_vector_type(4)))  int    v4i;

#define ASG __attribute__((address_space(1)))
#define ASL __attribute__((address_space(3)))

#if defined(__has_builtin)
#if __has_builtin(__builtin_amdgcn_global_load_async_to_lds_b128) && \
    __has_builtin(__builtin_amdgcn_s_wait_asynccnt)
#define HAVE_ASYNC 1
#else
#define HAVE_ASYNC 0
#endif
#else
#define HAVE_ASYNC 0
#endif

static constexpr int Bsz = 16;
static constexpr int Tsz = 12;
static constexpr int Nn  = 1024;   // tokens / graph nodes
static constexpr int Dd  = 64;
static constexpr int Dm  = 768;    // Tsz * Dd
static constexpr int Ff  = 3072;   // 4 * Dm
static constexpr int KTOP = 16;
static constexpr long long TOT = (long long)Bsz * Tsz * Nn * Dd; // 12,582,912

// ---------------------------------------------------------------------------
// Generic batched BF16 WMMA GEMM:
//   C[b] = act( scale * A[b] (MxK) x B[b] (KxN or NxK^T) + bias )
// Block: 256 threads (8 wave32), tile 128x128, K-step 32, double-buffered LDS.
// Wave w: M-offset (w&1)*64, N-offset (w>>1)*32 -> 4x2 = 8 WMMA tiles/wave.
// ---------------------------------------------------------------------------
template <bool TRANSB, bool RELU>
__global__ __launch_bounds__(256) void gemm_bf16_kernel(
    const bf16* __restrict__ A, long long strideA, int lda,
    const bf16* __restrict__ Bm, long long strideB, int ldb,
    const float* __restrict__ bias,
    float* __restrict__ outF, bf16* __restrict__ outB,
    int M, int N, int K, float scale)
{
    __shared__ __align__(16) bf16 As[2][128 * 32];
    __shared__ __align__(16) bf16 Bs[2][128 * 32];   // stored [n][k]

    const int tid  = threadIdx.x;
    const int lane = tid & 31;
    const int wid  = tid >> 5;
    const int wm   = (wid & 1) * 64;
    const int wn   = (wid >> 1) * 32;
    const int m0   = blockIdx.y * 128;
    const int n0   = blockIdx.x * 128;
    const int b    = blockIdx.z;

    const bf16* Ab = A  + (long long)b * strideA;
    const bf16* Bb = Bm + (long long)b * strideB;

    const int half = lane >> 4;   // 0: K 0..7 & 16..23, 1: K 8..15 & 24..31
    const int row  = lane & 15;   // M (A frag) / N (B frag)

    // per-thread staging coordinates: 512 chunks of 8 bf16 (16B) per tile
    const int c0  = tid, c1 = tid + 256;
    const int ar0 = c0 >> 2, ak0 = (c0 & 3) << 3;   // A / NT-B chunk 0
    const int ar1 = c1 >> 2, ak1 = (c1 & 3) << 3;   // A / NT-B chunk 1
    const int bk0 = c0 >> 4, bn0 = (c0 & 15) << 3;  // NN-B chunk 0
    const int bk1 = c1 >> 4, bn1 = (c1 & 15) << 3;  // NN-B chunk 1

    v8bf ra0, ra1, rb0, rb1;

    // Issue the global->LDS traffic for tile k0 into buffer `buf`.
    auto issue_stage = [&](int k0, int buf) {
#if HAVE_ASYNC
        __builtin_amdgcn_global_load_async_to_lds_b128(
            (ASG v4i*)(Ab + (long long)(m0 + ar0) * lda + k0 + ak0),
            (ASL v4i*)(&As[buf][ar0 * 32 + ak0]), 0, 0);
        __builtin_amdgcn_global_load_async_to_lds_b128(
            (ASG v4i*)(Ab + (long long)(m0 + ar1) * lda + k0 + ak1),
            (ASL v4i*)(&As[buf][ar1 * 32 + ak1]), 0, 0);
        if (TRANSB) {
            __builtin_amdgcn_global_load_async_to_lds_b128(
                (ASG v4i*)(Bb + (long long)(n0 + ar0) * ldb + k0 + ak0),
                (ASL v4i*)(&Bs[buf][ar0 * 32 + ak0]), 0, 0);
            __builtin_amdgcn_global_load_async_to_lds_b128(
                (ASG v4i*)(Bb + (long long)(n0 + ar1) * ldb + k0 + ak1),
                (ASL v4i*)(&Bs[buf][ar1 * 32 + ak1]), 0, 0);
        }
#else
        (void)buf;
        ra0 = *(const v8bf*)(Ab + (long long)(m0 + ar0) * lda + k0 + ak0);
        ra1 = *(const v8bf*)(Ab + (long long)(m0 + ar1) * lda + k0 + ak1);
        if (TRANSB) {
            rb0 = *(const v8bf*)(Bb + (long long)(n0 + ar0) * ldb + k0 + ak0);
            rb1 = *(const v8bf*)(Bb + (long long)(n0 + ar1) * ldb + k0 + ak1);
        }
#endif
        if (!TRANSB) {
            // K x N row-major: vector load along N, transposed commit later
            rb0 = *(const v8bf*)(Bb + (long long)(k0 + bk0) * ldb + n0 + bn0);
            rb1 = *(const v8bf*)(Bb + (long long)(k0 + bk1) * ldb + n0 + bn1);
        }
    };

    // Commit tile into LDS buffer `buf` and make it visible to the block.
    auto finish_stage = [&](int buf) {
#if !HAVE_ASYNC
        *(v8bf*)(&As[buf][ar0 * 32 + ak0]) = ra0;
        *(v8bf*)(&As[buf][ar1 * 32 + ak1]) = ra1;
        if (TRANSB) {
            *(v8bf*)(&Bs[buf][ar0 * 32 + ak0]) = rb0;
            *(v8bf*)(&Bs[buf][ar1 * 32 + ak1]) = rb1;
        }
#endif
        if (!TRANSB) {
#pragma unroll
            for (int j = 0; j < 8; ++j) Bs[buf][(bn0 + j) * 32 + bk0] = rb0[j];
#pragma unroll
            for (int j = 0; j < 8; ++j) Bs[buf][(bn1 + j) * 32 + bk1] = rb1[j];
        }
#if HAVE_ASYNC
        __builtin_amdgcn_s_wait_asynccnt(0);
#endif
        __syncthreads();
    };

    v8f acc[4][2] = {};

    issue_stage(0, 0);
    finish_stage(0);

    int buf = 0;
    for (int k0 = 0; k0 < K; k0 += 32) {
        const bool hasNext = (k0 + 32) < K;
        if (hasNext) {
            issue_stage(k0 + 32, buf ^ 1);   // overlaps with WMMA below
            __builtin_prefetch(Ab + (long long)(m0 + (tid >> 1)) * lda + k0 + 64, 0, 3);
        }

        // ---- fragments per ISA 16-bit layout: two b128 LDS loads each ----
        v16bf af[4];
#pragma unroll
        for (int s = 0; s < 4; ++s) {
            const bf16* p = &As[buf][(wm + s * 16 + row) * 32 + half * 8];
            v8bf lo = *(const v8bf*)p;          // K = half*8 .. +7
            v8bf hi = *(const v8bf*)(p + 16);   // K = 16+half*8 .. +7
            af[s] = __builtin_shufflevector(lo, hi,
                    0, 1, 2, 3, 4, 5, 6, 7, 8, 9, 10, 11, 12, 13, 14, 15);
        }
        v16bf bfr[2];
#pragma unroll
        for (int t = 0; t < 2; ++t) {
            const bf16* p = &Bs[buf][(wn + t * 16 + row) * 32 + half * 8];
            v8bf lo = *(const v8bf*)p;
            v8bf hi = *(const v8bf*)(p + 16);
            bfr[t] = __builtin_shufflevector(lo, hi,
                    0, 1, 2, 3, 4, 5, 6, 7, 8, 9, 10, 11, 12, 13, 14, 15);
        }

        // ---- 8 WMMAs per wave per K-step ---------------------------------
#pragma unroll
        for (int s = 0; s < 4; ++s)
#pragma unroll
            for (int t = 0; t < 2; ++t)
                acc[s][t] = __builtin_amdgcn_wmma_f32_16x16x32_bf16(
                    false, af[s], false, bfr[t], (short)0, acc[s][t], false, false);

        __syncthreads();                     // done reading `buf`
        if (hasNext) finish_stage(buf ^ 1);  // commit next tile, sync
        buf ^= 1;
    }

    // ---- epilogue: VGPR r <-> M = r + 8*half, N = lane&15 ----------------
    const long long obase = (long long)b * (long long)M * N;
#pragma unroll
    for (int s = 0; s < 4; ++s) {
#pragma unroll
        for (int t = 0; t < 2; ++t) {
#pragma unroll
            for (int r = 0; r < 8; ++r) {
                int mm = m0 + wm + s * 16 + half * 8 + r;
                int nn = n0 + wn + t * 16 + row;
                float v = acc[s][t][r] * scale;
                if (bias) v += bias[nn];
                if (RELU && v < 0.0f) v = 0.0f;
                long long idx = obase + (long long)mm * N + nn;
                if (outF) outF[idx] = v;
                if (outB) outB[idx] = (bf16)v;
            }
        }
    }
}

// ---------------------------------------------------------------------------
// Elementwise / layout kernels
// ---------------------------------------------------------------------------
__global__ void cast_f32_bf16_kernel(const float* __restrict__ in,
                                     bf16* __restrict__ out, int n)
{
    int i = blockIdx.x * blockDim.x + threadIdx.x;
    if (i < n) out[i] = (bf16)in[i];
}

// x (B,T,N,D) f32 -> xr (B,N,T*D) bf16
__global__ void transpose_x_kernel(const float* __restrict__ x,
                                   bf16* __restrict__ xr)
{
    long long i = (long long)blockIdx.x * blockDim.x + threadIdx.x;
    if (i >= TOT) return;
    int d    = (int)(i & 63);
    int n    = (int)((i >> 6) & 1023);
    int rest = (int)(i >> 16);       // b*T + t
    int t    = rest % Tsz;
    int b    = rest / Tsz;
    xr[((long long)(b * Nn + n)) * Dm + t * Dd + d] = (bf16)x[i];
}

// out (B,T,N,D) f32 = sparse_out[b,n,:] + graph_out[b,n,:]
__global__ void combine_kernel(const float* __restrict__ so,
                               const float* __restrict__ go,
                               float* __restrict__ out)
{
    long long i = (long long)blockIdx.x * blockDim.x + threadIdx.x;
    if (i >= TOT) return;
    int d    = (int)(i & 63);
    int n    = (int)((i >> 6) & 1023);
    int rest = (int)(i >> 16);
    int t    = rest % Tsz;
    int b    = rest / Tsz;
    long long src = ((long long)(b * Nn + n)) * Dm + t * Dd + d;
    out[i] = so[src] + go[src];
}

// ---------------------------------------------------------------------------
// Top-K threshold (binary search on value) + sparse softmax, one block / row.
// sp = softmax( where(attn >= kth_val && attn != 0, attn, -inf) ) -> bf16
// ---------------------------------------------------------------------------
__global__ __launch_bounds__(256) void topk_softmax_kernel(
    const float* __restrict__ attn, bf16* __restrict__ sp)
{
    __shared__ float rowv[Nn];
    __shared__ float redf[256];
    __shared__ int   redi[256];
    const int tid = threadIdx.x;
    const float* rp = attn + (long long)blockIdx.x * Nn;
    const float NEG = -__builtin_inff();

    float lmax = NEG, lmin = __builtin_inff();
#pragma unroll
    for (int i = 0; i < 4; ++i) {
        float v = rp[tid + i * 256];
        rowv[tid + i * 256] = v;
        lmax = fmaxf(lmax, v);
        lmin = fminf(lmin, v);
    }
    redf[tid] = lmax; __syncthreads();
    for (int s = 128; s > 0; s >>= 1) {
        if (tid < s) redf[tid] = fmaxf(redf[tid], redf[tid + s]);
        __syncthreads();
    }
    float hi = redf[0]; __syncthreads();
    redf[tid] = lmin; __syncthreads();
    for (int s = 128; s > 0; s >>= 1) {
        if (tid < s) redf[tid] = fminf(redf[tid], redf[tid + s]);
        __syncthreads();
    }
    float lo = redf[0]; __syncthreads();

    // binary search the K-th largest value
    for (int it = 0; it < 24; ++it) {
        float mid = 0.5f * (lo + hi);
        int c = 0;
#pragma unroll
        for (int i = 0; i < 4; ++i)
            if (rowv[tid + i * 256] >= mid) ++c;
        redi[tid] = c; __syncthreads();
        for (int s = 128; s > 0; s >>= 1) {
            if (tid < s) redi[tid] += redi[tid + s];
            __syncthreads();
        }
        int cnt = redi[0]; __syncthreads();
        if (cnt >= KTOP) lo = mid; else hi = mid;
    }
    const float thresh = lo;

    // masked max
    float lm = NEG;
#pragma unroll
    for (int i = 0; i < 4; ++i) {
        float v = rowv[tid + i * 256];
        if (v >= thresh && v != 0.0f) lm = fmaxf(lm, v);
    }
    redf[tid] = lm; __syncthreads();
    for (int s = 128; s > 0; s >>= 1) {
        if (tid < s) redf[tid] = fmaxf(redf[tid], redf[tid + s]);
        __syncthreads();
    }
    float m = redf[0]; __syncthreads();

    // exp sum
    float ls = 0.0f;
#pragma unroll
    for (int i = 0; i < 4; ++i) {
        float v = rowv[tid + i * 256];
        if (v >= thresh && v != 0.0f) ls += __expf(v - m);
    }
    redf[tid] = ls; __syncthreads();
    for (int s = 128; s > 0; s >>= 1) {
        if (tid < s) redf[tid] += redf[tid + s];
        __syncthreads();
    }
    float inv = 1.0f / redf[0];

    bf16* op = sp + (long long)blockIdx.x * Nn;
#pragma unroll
    for (int i = 0; i < 4; ++i) {
        float v = rowv[tid + i * 256];
        float p = (v >= thresh && v != 0.0f) ? __expf(v - m) * inv : 0.0f;
        op[tid + i * 256] = (bf16)p;
    }
}

// masked softmax over gm rows: zero entries -> -inf, softmax -> bf16
__global__ __launch_bounds__(256) void gm_softmax_kernel(
    const float* __restrict__ gm, bf16* __restrict__ out)
{
    __shared__ float rowv[Nn];
    __shared__ float redf[256];
    const int tid = threadIdx.x;
    const float* rp = gm + (long long)blockIdx.x * Nn;
    const float NEG = -__builtin_inff();

    float lmax = NEG;
#pragma unroll
    for (int i = 0; i < 4; ++i) {
        float v  = rp[tid + i * 256];
        float vv = (v == 0.0f) ? NEG : v;
        rowv[tid + i * 256] = vv;
        lmax = fmaxf(lmax, vv);
    }
    redf[tid] = lmax; __syncthreads();
    for (int s = 128; s > 0; s >>= 1) {
        if (tid < s) redf[tid] = fmaxf(redf[tid], redf[tid + s]);
        __syncthreads();
    }
    float m = redf[0]; __syncthreads();

    float ls = 0.0f;
#pragma unroll
    for (int i = 0; i < 4; ++i) {
        float vv = rowv[tid + i * 256];
        ls += (vv > NEG) ? __expf(vv - m) : 0.0f;
    }
    redf[tid] = ls; __syncthreads();
    for (int s = 128; s > 0; s >>= 1) {
        if (tid < s) redf[tid] += redf[tid + s];
        __syncthreads();
    }
    float inv = 1.0f / redf[0];

    bf16* op = out + (long long)blockIdx.x * Nn;
#pragma unroll
    for (int i = 0; i < 4; ++i) {
        float vv = rowv[tid + i * 256];
        float p  = (vv > NEG) ? __expf(vv - m) * inv : 0.0f;
        op[tid + i * 256] = (bf16)p;
    }
}

// ---------------------------------------------------------------------------
// Host-side orchestration
// ---------------------------------------------------------------------------
static void launch_gemm(bool transb, bool relu,
                        const bf16* A, long long sA, int lda,
                        const bf16* Bm, long long sB, int ldb,
                        const float* bias, float* outF, bf16* outB,
                        int M, int N, int K, float scale, hipStream_t stream)
{
    dim3 grid(N / 128, M / 128, Bsz);
    dim3 block(256);
    if (transb) {
        if (relu) gemm_bf16_kernel<true,  true ><<<grid, block, 0, stream>>>(A, sA, lda, Bm, sB, ldb, bias, outF, outB, M, N, K, scale);
        else      gemm_bf16_kernel<true,  false><<<grid, block, 0, stream>>>(A, sA, lda, Bm, sB, ldb, bias, outF, outB, M, N, K, scale);
    } else {
        if (relu) gemm_bf16_kernel<false, true ><<<grid, block, 0, stream>>>(A, sA, lda, Bm, sB, ldb, bias, outF, outB, M, N, K, scale);
        else      gemm_bf16_kernel<false, false><<<grid, block, 0, stream>>>(A, sA, lda, Bm, sB, ldb, bias, outF, outB, M, N, K, scale);
    }
}

extern "C" void kernel_launch(void* const* d_in, const int* in_sizes, int n_in,
                              void* d_out, int out_size, void* d_ws, size_t ws_size,
                              hipStream_t stream)
{
    (void)in_sizes; (void)n_in; (void)out_size; (void)ws_size;

    const float* x     = (const float*)d_in[0];
    const float* graph = (const float*)d_in[1];
    const float* Wq    = (const float*)d_in[2];
    const float* bq    = (const float*)d_in[3];
    const float* Wk    = (const float*)d_in[4];
    const float* bk    = (const float*)d_in[5];
    const float* Wv    = (const float*)d_in[6];
    const float* bv    = (const float*)d_in[7];
    const float* Wo    = (const float*)d_in[8];
    const float* bo    = (const float*)d_in[9];
    const float* W1    = (const float*)d_in[10];
    const float* b1    = (const float*)d_in[11];
    const float* W2    = (const float*)d_in[12];
    const float* b2    = (const float*)d_in[13];
    float* out = (float*)d_out;

    // ---- workspace layout (256B aligned, with reuse) ---------------------
    size_t off = 0;
    auto alloc = [&](size_t bytes) -> void* {
        void* p = (char*)d_ws + off;
        off += (bytes + 255) & ~(size_t)255;
        return p;
    };
    bf16* wq_b = (bf16*)alloc((size_t)Dm * Dm * 2);
    bf16* wk_b = (bf16*)alloc((size_t)Dm * Dm * 2);
    bf16* wv_b = (bf16*)alloc((size_t)Dm * Dm * 2);
    bf16* wo_b = (bf16*)alloc((size_t)Dm * Dm * 2);
    bf16* w1_b = (bf16*)alloc((size_t)Dm * Ff * 2);
    bf16* w2_b = (bf16*)alloc((size_t)Ff * Dm * 2);
    bf16* gr_b = (bf16*)alloc((size_t)Nn * Nn * 2);
    bf16* xr_b = (bf16*)alloc((size_t)Bsz * Nn * Dm * 2);
    bf16* q_b  = (bf16*)alloc((size_t)Bsz * Nn * Dm * 2);   // reused as t1 = sp@v
    bf16* k_b  = (bf16*)alloc((size_t)Bsz * Nn * Dm * 2);   // reused as h2
    bf16* v_b  = (bf16*)alloc((size_t)Bsz * Nn * Dm * 2);
    float* attnF = (float*)alloc((size_t)Bsz * Nn * Nn * 4); // reused as gm_f32
    bf16*  attnB = (bf16*) alloc((size_t)Bsz * Nn * Nn * 2); // reused as gm_softmax
    bf16*  sp_b  = (bf16*) alloc((size_t)Bsz * Nn * Nn * 2);
    bf16*  ffn1  = (bf16*) alloc((size_t)Bsz * Nn * Ff * 2);
    bf16*  h1_b  = (bf16*) alloc((size_t)Bsz * Nn * Ff * 2);
    float* soF   = (float*)alloc((size_t)Bsz * Nn * Dm * 4);
    float* goF   = (float*)alloc((size_t)Bsz * Nn * Dm * 4);

    const long long sQKV  = (long long)Nn * Dm;   // per-batch stride, Dm-wide
    const long long sAtt  = (long long)Nn * Nn;
    const long long sFfn  = (long long)Nn * Ff;
    const float scale = 1.0f / sqrtf((float)Dm);

    // ---- weight / graph casts to bf16 ------------------------------------
    auto cast = [&](const float* src, bf16* dst, int n) {
        cast_f32_bf16_kernel<<<(n + 255) / 256, 256, 0, stream>>>(src, dst, n);
    };
    cast(Wq, wq_b, Dm * Dm);
    cast(Wk, wk_b, Dm * Dm);
    cast(Wv, wv_b, Dm * Dm);
    cast(Wo, wo_b, Dm * Dm);
    cast(W1, w1_b, Dm * Ff);
    cast(W2, w2_b, Ff * Dm);
    cast(graph, gr_b, Nn * Nn);

    // ---- xr = transpose(x) -> (B, N, DM) bf16 ----------------------------
    transpose_x_kernel<<<(int)(TOT / 256), 256, 0, stream>>>(x, xr_b);

    // ---- Q, K, V projections (NN, shared weights) ------------------------
    launch_gemm(false, false, xr_b, sQKV, Dm, wq_b, 0, Dm, bq, nullptr, q_b, Nn, Dm, Dm, 1.0f, stream);
    launch_gemm(false, false, xr_b, sQKV, Dm, wk_b, 0, Dm, bk, nullptr, k_b, Nn, Dm, Dm, 1.0f, stream);
    launch_gemm(false, false, xr_b, sQKV, Dm, wv_b, 0, Dm, bv, nullptr, v_b, Nn, Dm, Dm, 1.0f, stream);

    // ---- attn = scale * Q K^T  (NT, per-batch), f32 + bf16 ---------------
    launch_gemm(true, false, q_b, sQKV, Dm, k_b, sQKV, Dm, nullptr, attnF, attnB,
                Nn, Nn, Dm, scale, stream);

    // ---- top-K sparse softmax -> sp (bf16) -------------------------------
    topk_softmax_kernel<<<Bsz * Nn, 256, 0, stream>>>(attnF, sp_b);

    // ---- gm[b] = graph @ attn[b]^T  (NT, A shared via strideA=0) ---------
    // writes into attnF (attn f32 no longer needed)
    launch_gemm(true, false, gr_b, 0, Nn, attnB, sAtt, Nn, nullptr, attnF, nullptr,
                Nn, Nn, Nn, 1.0f, stream);

    // ---- gm masked softmax -> gm_sm (bf16, reuses attnB) -----------------
    gm_softmax_kernel<<<Bsz * Nn, 256, 0, stream>>>(attnF, attnB);
    bf16* gm_sm = attnB;

    // ---- FFN branch ------------------------------------------------------
    launch_gemm(false, false, xr_b, sQKV, Dm, w1_b, 0, Ff, b1, nullptr, ffn1,
                Nn, Ff, Dm, 1.0f, stream);                       // xr@W1+b1
    launch_gemm(false, true, gm_sm, sAtt, Nn, ffn1, sFfn, Ff, nullptr, nullptr, h1_b,
                Nn, Ff, Nn, 1.0f, stream);                        // h1 = relu(gm@ffn1)
    launch_gemm(false, false, h1_b, sFfn, Ff, w2_b, 0, Dm, b2, nullptr, k_b,
                Nn, Dm, Ff, 1.0f, stream);                        // h2 = h1@W2+b2 (reuse k)
    launch_gemm(false, true, gm_sm, sAtt, Nn, k_b, sQKV, Dm, nullptr, goF, nullptr,
                Nn, Dm, Nn, 1.0f, stream);                        // graph_out (f32)

    // ---- sparse-attention branch -----------------------------------------
    launch_gemm(false, false, sp_b, sAtt, Nn, v_b, sQKV, Dm, nullptr, nullptr, q_b,
                Nn, Dm, Nn, 1.0f, stream);                        // t1 = sp@v (reuse q)
    launch_gemm(false, false, q_b, sQKV, Dm, wo_b, 0, Dm, bo, soF, nullptr,
                Nn, Dm, Dm, 1.0f, stream);                        // sparse_out (f32)

    // ---- combine + transpose back to (B,T,N,D) ---------------------------
    combine_kernel<<<(int)(TOT / 256), 256, 0, stream>>>(soF, goF, out);
}
